// PointNet2MSG_2156073583005
// MI455X (gfx1250) — compile-verified, matrix operand order checked by reading, not run on hardware
//
#include <hip/hip_runtime.h>

typedef _Float16 h16;
typedef __attribute__((ext_vector_type(16))) _Float16 v16h;
typedef __attribute__((ext_vector_type(8)))  _Float16 h8;
typedef __attribute__((ext_vector_type(8)))  float    v8f;

// ---------------------------------------------------------------------------
// split pointcloud (B,N,4) -> xyz (B,N,3), feats0 (B,N,1)
// ---------------------------------------------------------------------------
__global__ void split_pc_kernel(const float* __restrict__ pc, float* __restrict__ xyz,
                                float* __restrict__ f0, int total) {
  int i = blockIdx.x * blockDim.x + threadIdx.x;
  if (i >= total) return;
  xyz[3*i+0] = pc[4*i+0];
  xyz[3*i+1] = pc[4*i+1];
  xyz[3*i+2] = pc[4*i+2];
  f0[i]      = pc[4*i+3];
}

// ---------------------------------------------------------------------------
// Farthest point sampling: one block per batch, LDS tree argmax per iteration.
// Matches reference: fidx[0]=0; centroid=xyz[far]; dist=min(dist,d); far=argmax.
// ---------------------------------------------------------------------------
__global__ void fps_kernel(const float* __restrict__ xyz, float* __restrict__ dist,
                           int* __restrict__ fidx, int Npts, int npoint) {
  const int b   = blockIdx.x;
  const int tid = threadIdx.x;
  const float* xb = xyz + (size_t)b * Npts * 3;
  float* db = dist + (size_t)b * Npts;
  int*   fb = fidx + (size_t)b * npoint;
  for (int n = tid; n < Npts; n += blockDim.x) db[n] = 1e10f;
  __shared__ float sd[256];
  __shared__ int   si[256];
  __shared__ int   sfar;
  if (tid == 0) sfar = 0;
  __syncthreads();
  for (int it = 0; it < npoint; ++it) {
    const int far = sfar;
    if (tid == 0) fb[it] = far;
    const float cx = xb[3*far], cy = xb[3*far+1], cz = xb[3*far+2];
    float bestd = -1.0f; int besti = Npts;
    for (int n = tid; n < Npts; n += blockDim.x) {
      float dx = xb[3*n] - cx, dy = xb[3*n+1] - cy, dz = xb[3*n+2] - cz;
      float d  = dx*dx + dy*dy + dz*dz;
      float dm = fminf(db[n], d);
      db[n] = dm;
      if (dm > bestd) { bestd = dm; besti = n; }   // strict > keeps first index
    }
    sd[tid] = bestd; si[tid] = besti;
    __syncthreads();
    for (int s = 128; s > 0; s >>= 1) {
      if (tid < s) {
        float od = sd[tid+s]; int oi = si[tid+s];
        if (od > sd[tid] || (od == sd[tid] && oi < si[tid])) { sd[tid] = od; si[tid] = oi; }
      }
      __syncthreads();
    }
    if (tid == 0) sfar = si[0];
    __syncthreads();
  }
}

// ---------------------------------------------------------------------------
// Gather sampled centroids
// ---------------------------------------------------------------------------
__global__ void gather_xyz_kernel(const float* __restrict__ xyz, const int* __restrict__ fidx,
                                  float* __restrict__ nxyz, int Nin, int npt, int total) {
  int i = blockIdx.x * blockDim.x + threadIdx.x;
  if (i >= total) return;           // i = b*npt + s
  int b = i / npt;
  int id = fidx[i];
  const float* s = xyz + ((size_t)b * Nin + id) * 3;
  nxyz[3*i] = s[0]; nxyz[3*i+1] = s[1]; nxyz[3*i+2] = s[2];
}

// ---------------------------------------------------------------------------
// Ball query: first ns in-radius indices in ascending order; pad with first
// in-radius index; all-zeros if none (matches reference semantics).
// ---------------------------------------------------------------------------
__global__ void ball_query_kernel(const float* __restrict__ xyz, const float* __restrict__ nxyz,
                                  int* __restrict__ bidx, int Nin, int npt, int ns,
                                  float r2, int total) {
  int i = blockIdx.x * blockDim.x + threadIdx.x;
  if (i >= total) return;           // i = b*npt + s
  int b = i / npt;
  const float qx = nxyz[3*i], qy = nxyz[3*i+1], qz = nxyz[3*i+2];
  const float* xb = xyz + (size_t)b * Nin * 3;
  int* out = bidx + (size_t)i * ns;
  int cnt = 0, first = -1;
  for (int n = 0; n < Nin && cnt < ns; ++n) {
    float dx = xb[3*n] - qx, dy = xb[3*n+1] - qy, dz = xb[3*n+2] - qz;
    float d = dx*dx + dy*dy + dz*dz;
    if (d < r2) { if (first < 0) first = n; out[cnt++] = n; }
  }
  int fill = (first < 0) ? 0 : first;
  for (int k = cnt; k < ns; ++k) out[k] = fill;
}

// ---------------------------------------------------------------------------
// Build grouped GEMM input A[M, Kp] f16: [xyz-centered | feats | 0-pad]
// ---------------------------------------------------------------------------
__global__ void build_group_kernel(const float* __restrict__ xyz, const float* __restrict__ feats,
                                   const float* __restrict__ nxyz, const int* __restrict__ bidx,
                                   h16* __restrict__ A, int Nin, int npt, int ns, int Cf,
                                   int Kp, long long total) {
  long long t = (long long)blockIdx.x * blockDim.x + threadIdx.x;
  if (t >= total) return;
  int c = (int)(t % Kp);
  long long row = t / Kp;           // (b*npt + s)*ns + k
  long long bs  = row / ns;         // b*npt + s
  int b = (int)(bs / npt);
  int id = bidx[row];
  float v = 0.0f;
  if (c < 3)            v = xyz[((size_t)b * Nin + id) * 3 + c] - nxyz[bs * 3 + c];
  else if (c < 3 + Cf)  v = feats[((size_t)b * Nin + id) * Cf + (c - 3)];
  A[t] = (h16)v;
}

// ---------------------------------------------------------------------------
// Stage weights W(Nout,K) f32 -> Wt(Np,Kp) f16 zero-padded; bias -> padded f32
// ---------------------------------------------------------------------------
__global__ void stage_weights_kernel(const float* __restrict__ W, const float* __restrict__ b,
                                     h16* __restrict__ Wt, float* __restrict__ bp,
                                     int Nout, int K, int Np, int Kp) {
  int t = blockIdx.x * blockDim.x + threadIdx.x;
  if (t >= Np * Kp) return;
  int k = t % Kp, n = t / Kp;
  float v = (n < Nout && k < K) ? W[(size_t)n * K + k] : 0.0f;
  Wt[t] = (h16)v;
  if (k == 0) bp[n] = (n < Nout) ? b[n] : 0.0f;
}

// ---------------------------------------------------------------------------
// Register-blocked WMMA GEMM + bias + ReLU.
// One wave (32 threads) computes a 64(M) x 32(N) macro-tile = 4x2 WMMA tiles.
// Per K=32 step: 4 A-fragments + 2 B-fragments (12 x b128 loads) feed 8
// v_wmma_f32_16x16x32_f16 ops -> 1.5 loads/wmma (vs 4 for unblocked).
// A: [M, Kp] f16 row-major (Kp % 32 == 0, M % 64 == 0)
// Wt: [Np, Kp] f16 (B^T layout, Np % 32 == 0)
// Fragment layout per ISA 7.12.2: lane%16 = row/col, lane/16 picks K-half,
// element e -> K = kbase + (e<8 ? e : e+8).  Epilogue per D layout:
// VGPR q, lanes 0-15 -> M=q, lanes 16-31 -> M=q+8.
// ---------------------------------------------------------------------------
#define MBLK 4
#define NBLK 2
__global__ void wmma_gemm_bias_relu(const h16* __restrict__ A, const h16* __restrict__ Wt,
                                    const float* __restrict__ bias,
                                    h16* __restrict__ outH, float* __restrict__ outF,
                                    int M, int Kp, int Np) {
  const int ntn  = Np >> 5;                  // macro-tiles along N (32 wide)
  const int tile = blockIdx.x;
  const int tm   = tile / ntn;               // macro-tile row (64 tall)
  const int tn   = tile - tm * ntn;
  const int lane = threadIdx.x;
  const int r    = lane & 15;
  const int hl   = lane >> 4;

  const h16* Ar[MBLK];
  const h16* Br[NBLK];
  #pragma unroll
  for (int j = 0; j < MBLK; ++j)
    Ar[j] = A + (size_t)(tm * 64 + j * 16 + r) * Kp + hl * 8;
  #pragma unroll
  for (int i = 0; i < NBLK; ++i)
    Br[i] = Wt + (size_t)(tn * 32 + i * 16 + r) * Kp + hl * 8;

  v8f c[MBLK][NBLK] = {};
  for (int k0 = 0; k0 < Kp; k0 += 32) {
    union { v16h v; h8 h[2]; } ua[MBLK], ub[NBLK];
    #pragma unroll
    for (int j = 0; j < MBLK; ++j) {
      ua[j].h[0] = *(const h8*)(Ar[j] + k0);
      ua[j].h[1] = *(const h8*)(Ar[j] + k0 + 16);
    }
    #pragma unroll
    for (int i = 0; i < NBLK; ++i) {
      ub[i].h[0] = *(const h8*)(Br[i] + k0);
      ub[i].h[1] = *(const h8*)(Br[i] + k0 + 16);
    }
    __builtin_prefetch(Ar[0] + k0 + 64, 0, 1);   // global_prefetch_b8 hint
    #pragma unroll
    for (int j = 0; j < MBLK; ++j)
      #pragma unroll
      for (int i = 0; i < NBLK; ++i)
        c[j][i] = __builtin_amdgcn_wmma_f32_16x16x32_f16(
            /*neg_a=*/false, ua[j].v, /*neg_b=*/false, ub[i].v,
            /*c_mod=*/(short)0, c[j][i], /*reuse_a=*/false, /*reuse_b=*/false);
  }

  const int mo = hl * 8;                      // D layout: lanes 16-31 -> M+8
  #pragma unroll
  for (int i = 0; i < NBLK; ++i) {
    const int n  = tn * 32 + i * 16 + r;
    const float bv = bias[n];
    #pragma unroll
    for (int j = 0; j < MBLK; ++j) {
      #pragma unroll
      for (int q = 0; q < 8; ++q) {
        float v = c[j][i][q] + bv;
        v = v > 0.0f ? v : 0.0f;
        size_t off = (size_t)(tm * 64 + j * 16 + mo + q) * Np + (size_t)n;
        outH[off] = (h16)v;
        if (outF) outF[off] = v;
      }
    }
  }
}

// ---------------------------------------------------------------------------
// Max over nsample axis -> concat into level feature buffer
// ---------------------------------------------------------------------------
__global__ void sa_maxpool_kernel(const h16* __restrict__ act, float* __restrict__ out,
                                  int npt, int ns, int Cout, int Np, int Ctot, int choff,
                                  int total) {
  int i = blockIdx.x * blockDim.x + threadIdx.x;
  if (i >= total) return;           // i = (b*npt+s)*Cout + c
  int c  = i % Cout;
  int bs = i / Cout;
  const h16* a = act + (size_t)bs * ns * Np + c;
  float m = -3e38f;
  for (int k = 0; k < ns; ++k) m = fmaxf(m, (float)a[(size_t)k * Np]);
  out[(size_t)bs * Ctot + choff + c] = m;
}

// ---------------------------------------------------------------------------
// 3-NN (squared dist) + inverse-distance weights (first-index tie-break)
// ---------------------------------------------------------------------------
__global__ void three_nn_kernel(const float* __restrict__ x1, const float* __restrict__ x2,
                                int N1, int N2, int* __restrict__ nnidx,
                                float* __restrict__ nnw, int total) {
  int i = blockIdx.x * blockDim.x + threadIdx.x;
  if (i >= total) return;           // i = b*N1 + n
  int b = i / N1;
  float px = x1[3*i], py = x1[3*i+1], pz = x1[3*i+2];
  const float* xb = x2 + (size_t)b * N2 * 3;
  float d0 = 3e38f, d1 = 3e38f, d2v = 3e38f;
  int   i0 = 0, i1 = 0, i2 = 0;
  for (int j = 0; j < N2; ++j) {
    float dx = xb[3*j] - px, dy = xb[3*j+1] - py, dz = xb[3*j+2] - pz;
    float d = dx*dx + dy*dy + dz*dz;
    if (d < d0)       { d2v = d1; i2 = i1; d1 = d0; i1 = i0; d0 = d; i0 = j; }
    else if (d < d1)  { d2v = d1; i2 = i1; d1 = d;  i1 = j; }
    else if (d < d2v) { d2v = d;  i2 = j; }
  }
  float w0 = 1.0f / (fmaxf(d0,  0.0f) + 1e-8f);
  float w1 = 1.0f / (fmaxf(d1,  0.0f) + 1e-8f);
  float w2 = 1.0f / (fmaxf(d2v, 0.0f) + 1e-8f);
  float s = w0 + w1 + w2;
  nnidx[3*i] = i0; nnidx[3*i+1] = i1; nnidx[3*i+2] = i2;
  nnw[3*i] = w0 / s; nnw[3*i+1] = w1 / s; nnw[3*i+2] = w2 / s;
}

// ---------------------------------------------------------------------------
// Build FP GEMM input: [interp(C2) | feats1(C1) | 0-pad] -> f16
// ---------------------------------------------------------------------------
__global__ void build_fp_A_kernel(const float* __restrict__ f2, const float* __restrict__ f1,
                                  const int* __restrict__ nnidx, const float* __restrict__ nnw,
                                  h16* __restrict__ A, int N1, int N2, int C2, int C1,
                                  int Kp, long long total) {
  long long t = (long long)blockIdx.x * blockDim.x + threadIdx.x;
  if (t >= total) return;
  int c = (int)(t % Kp);
  long long i = t / Kp;             // b*N1 + n
  int b = (int)(i / N1);
  float v = 0.0f;
  if (c < C2) {
    const float* fb = f2 + (size_t)b * N2 * C2;
    v = nnw[3*i]   * fb[(size_t)nnidx[3*i]   * C2 + c]
      + nnw[3*i+1] * fb[(size_t)nnidx[3*i+1] * C2 + c]
      + nnw[3*i+2] * fb[(size_t)nnidx[3*i+2] * C2 + c];
  } else if (c < C2 + C1) {
    v = f1[(size_t)i * C1 + (c - C2)];
  }
  A[t] = (h16)v;
}

// ---------------------------------------------------------------------------
// Final global max over N -> (B, C) output
// ---------------------------------------------------------------------------
__global__ void final_max_kernel(const float* __restrict__ pf0, float* __restrict__ out,
                                 int N, int C) {
  int i = threadIdx.x;              // one block of 256: i = b*C + c
  if (i >= 2 * C) return;
  int b = i / C, c = i % C;
  const float* p = pf0 + (size_t)b * N * C + c;
  float m = -3e38f;
  for (int n = 0; n < N; ++n) m = fmaxf(m, p[(size_t)n * C]);
  out[i] = m;
}

// ---------------------------------------------------------------------------
// Host orchestration
// ---------------------------------------------------------------------------
static inline int rnd32(int x) { return (x + 31) & ~31; }
static inline int cdiv(long long a, long long b) { return (int)((a + b - 1) / b); }

static void gemm_layer(const float* W, const float* bsrc, int K, int Nout,
                       const h16* Ain, h16* outH, float* outF, int M,
                       h16* wts, float* bp, hipStream_t stream) {
  int Kp = rnd32(K), Np = rnd32(Nout);
  stage_weights_kernel<<<cdiv((long long)Np * Kp, 256), 256, 0, stream>>>(
      W, bsrc, wts, bp, Nout, K, Np, Kp);
  int tiles = (M / 64) * (Np / 32);          // 64x32 macro-tile per wave
  wmma_gemm_bias_relu<<<tiles, 32, 0, stream>>>(Ain, wts, bp, outH, outF, M, Kp, Np);
}

extern "C" void kernel_launch(void* const* d_in, const int* in_sizes, int n_in,
                              void* d_out, int out_size, void* d_ws, size_t ws_size,
                              hipStream_t stream) {
  (void)in_sizes; (void)n_in; (void)out_size; (void)ws_size;

  const int NL[5]     = {16384, 4096, 1024, 256, 64};
  const int CL[5]     = {1, 96, 256, 512, 1024};          // pre-FP feature channels
  const int PFC[4]    = {128, 256, 512, 512};             // post-FP feature channels
  const int SA_NPT[4] = {4096, 1024, 256, 64};
  const float SA_R[4][2] = {{0.1f,0.5f},{0.5f,1.0f},{1.0f,2.0f},{2.0f,4.0f}};
  const int SA_NS[2]  = {16, 32};
  const int SA_DIMS[4][2][4] = {
      {{4,16,16,32},      {4,32,32,64}},
      {{99,64,64,128},    {99,64,96,128}},
      {{259,128,196,256}, {259,128,196,256}},
      {{515,256,256,512}, {515,256,384,512}}};
  const int FP_DIMS[4][3] = {{257,128,128},{608,256,256},{768,512,512},{1536,512,512}};

  // -------- workspace bump allocator --------
  char* base = (char*)d_ws;
  size_t cur = 0;
  auto alloc = [&](size_t bytes) -> char* {
    char* p = base + cur;
    cur += (bytes + 255) & ~(size_t)255;
    return p;
  };

  float* xyz[5]; float* feats[5]; float* pf[4]; int* fidx[4];
  for (int k = 0; k < 5; ++k) xyz[k]   = (float*)alloc((size_t)2 * NL[k] * 3 * sizeof(float));
  for (int k = 0; k < 5; ++k) feats[k] = (float*)alloc((size_t)2 * NL[k] * CL[k] * sizeof(float));
  for (int k = 0; k < 4; ++k) pf[k]    = (float*)alloc((size_t)2 * NL[k] * PFC[k] * sizeof(float));
  for (int k = 0; k < 4; ++k) fidx[k]  = (int*)alloc((size_t)2 * SA_NPT[k] * sizeof(int));
  float* distws = (float*)alloc((size_t)2 * 16384 * sizeof(float));
  int*   bidx   = (int*)alloc((size_t)2 * 4096 * 32 * sizeof(int));
  int*   nnidx  = (int*)alloc((size_t)2 * 16384 * 3 * sizeof(int));
  float* nnw    = (float*)alloc((size_t)2 * 16384 * 3 * sizeof(float));
  h16*   wts    = (h16*)alloc((size_t)(4u << 20) * sizeof(h16));   // 8 MB staged weights
  float* bp     = (float*)alloc((size_t)1024 * sizeof(float));
  h16*   ACT0   = (h16*)alloc((size_t)(20u << 20) * sizeof(h16));  // 40 MB ping
  h16*   ACT1   = (h16*)alloc((size_t)(20u << 20) * sizeof(h16));  // 40 MB pong

  const float* PC = (const float*)d_in[0];
  split_pc_kernel<<<cdiv(2 * 16384, 256), 256, 0, stream>>>(PC, xyz[0], feats[0], 2 * 16384);

  // -------- Set-abstraction levels --------
  for (int k = 0; k < 4; ++k) {
    const int Nin = NL[k], npt = SA_NPT[k];
    fps_kernel<<<2, 256, 0, stream>>>(xyz[k], distws, fidx[k], Nin, npt);
    gather_xyz_kernel<<<cdiv(2 * npt, 256), 256, 0, stream>>>(xyz[k], fidx[k], xyz[k+1],
                                                              Nin, npt, 2 * npt);
    const int Ctot = CL[k+1];
    int choff = 0;
    for (int s = 0; s < 2; ++s) {
      const int ns = SA_NS[s];
      const float r = SA_R[k][s];
      ball_query_kernel<<<cdiv(2 * npt, 256), 256, 0, stream>>>(
          xyz[k], xyz[k+1], bidx, Nin, npt, ns, r * r, 2 * npt);
      const int K0 = SA_DIMS[k][s][0], Kp0 = rnd32(K0);
      const long long M = 2LL * npt * ns;
      build_group_kernel<<<cdiv(M * Kp0, 256), 256, 0, stream>>>(
          xyz[k], feats[k], xyz[k+1], bidx, ACT0, Nin, npt, ns, CL[k], Kp0, M * Kp0);
      const h16* src = ACT0;
      h16* dst = ACT1;
      int Nlast = 0;
      for (int l = 0; l < 3; ++l) {
        const int Kin = SA_DIMS[k][s][l], Nout = SA_DIMS[k][s][l+1];
        const int wi = 1 + k * 12 + s * 6 + l * 2;
        gemm_layer((const float*)d_in[wi], (const float*)d_in[wi + 1], Kin, Nout,
                   src, dst, nullptr, (int)M, wts, bp, stream);
        const h16* t = src; src = dst; dst = (h16*)t;
        Nlast = Nout;
      }
      sa_maxpool_kernel<<<cdiv(2LL * npt * Nlast, 256), 256, 0, stream>>>(
          src, feats[k+1], npt, ns, Nlast, rnd32(Nlast), Ctot, choff,
          (int)(2LL * npt * Nlast));
      choff += Nlast;
    }
  }

  // -------- Feature-propagation levels (i = 3 .. 0) --------
  for (int i = 3; i >= 0; --i) {
    const int N1 = NL[i], N2 = NL[i+1];
    const float* f2 = (i == 3) ? feats[4] : pf[i+1];
    const int C2    = (i == 3) ? CL[4]    : PFC[i+1];
    const float* f1 = feats[i];
    const int C1    = CL[i];
    three_nn_kernel<<<cdiv(2 * N1, 256), 256, 0, stream>>>(xyz[i], xyz[i+1], N1, N2,
                                                           nnidx, nnw, 2 * N1);
    const int Kp0 = rnd32(FP_DIMS[i][0]);           // FP_DIMS[i][0] == C2 + C1
    const long long tot = 2LL * N1 * Kp0;
    build_fp_A_kernel<<<cdiv(tot, 256), 256, 0, stream>>>(f2, f1, nnidx, nnw, ACT0,
                                                          N1, N2, C2, C1, Kp0, tot);
    const int M  = 2 * N1;
    const int wi = 49 + i * 4;
    gemm_layer((const float*)d_in[wi],     (const float*)d_in[wi + 1],
               FP_DIMS[i][0], FP_DIMS[i][1], ACT0, ACT1, nullptr, M, wts, bp, stream);
    gemm_layer((const float*)d_in[wi + 2], (const float*)d_in[wi + 3],
               FP_DIMS[i][1], FP_DIMS[i][2], ACT1, ACT0, pf[i], M, wts, bp, stream);
  }

  final_max_kernel<<<1, 256, 0, stream>>>(pf[0], (float*)d_out, NL[0], PFC[0]);
}
